// DNA_49289044689248
// MI455X (gfx1250) — compile-verified, matrix operand order checked by reading, not run on hardware
//
#include <hip/hip_runtime.h>
#include <cmath>

typedef __attribute__((ext_vector_type(16))) _Float16 v16h;
typedef __attribute__((ext_vector_type(8)))  _Float16 v8h;
typedef __attribute__((ext_vector_type(8)))  float    v8f;

#define HID 64
#define HEADS 8
#define LDS3 __attribute__((address_space(3)))

// ---------------------------------------------------------------------------
// WMMA 16-bit fragment K packing (ISA 7.12.2): per lane, halves live at
//   K = kbase+0..7   (VGPR 0-3)  and  K = 16+kbase+0..7 (VGPR 4-7),
// kbase = (lane&16) ? 8 : 0.  Two contiguous 16B runs -> two b128 loads.
// ---------------------------------------------------------------------------
__device__ __forceinline__ int koff_fn(int v, int kbase) {
  return (v < 4) ? (kbase + 2 * v) : (16 + kbase + 2 * (v - 4));
}

__device__ __forceinline__ void atomic_add_f32(float* p, float v) {
  __hip_atomic_fetch_add(p, v, __ATOMIC_RELAXED, __HIP_MEMORY_SCOPE_AGENT);
}

// ---------------------------------------------------------------------------
// Pre-pack a f32 weight matrix W[K x 64] into f16 WMMA-B fragment order:
//   out[((ks*4 + nt)*32 + lane)*16 + j],  j = 2v+s -> K = ks*32+koff(v)+s,
//   N = nt*16 + (lane&15).  One half per thread.
// ---------------------------------------------------------------------------
__global__ void pack_w_kernel(const float* __restrict__ W,
                              _Float16* __restrict__ out, int K) {
  int gid = blockIdx.x * blockDim.x + threadIdx.x;
  if (gid >= K * 64) return;
  const int j    = gid & 15;
  const int lane = (gid >> 4) & 31;
  const int nt   = (gid >> 9) & 3;
  const int ks   = gid >> 11;
  const int kbase = (lane & 16) ? 8 : 0;
  const int kk = ks * 32 + koff_fn(j >> 1, kbase) + (j & 1);
  const int n  = nt * 16 + (lane & 15);
  out[gid] = (_Float16)W[(size_t)kk * HID + n];
}

__global__ void f32_to_f16_kernel(const float* __restrict__ src,
                                  _Float16* __restrict__ dst, int n) {
  int i = blockIdx.x * blockDim.x + threadIdx.x;
  if (i < n) dst[i] = (_Float16)src[i];
}

// ---------------------------------------------------------------------------
// C[M x 64] = act(A[M x K] @ W + bias).  A: f16 row-major.  Wpk: f16 packed
// fragments (global), staged to LDS via async global->LDS b128 (ASYNCcnt).
// One wave per 16-row tile, 4 waves/block, 4 WMMAs per 32-wide k-step.
// ---------------------------------------------------------------------------
template <int K, bool RELU, typename OT>
__global__ __launch_bounds__(128) void wmma_gemm_kernel(
    const _Float16* __restrict__ A, const _Float16* __restrict__ Wpk,
    const float* __restrict__ bias, OT* __restrict__ C, int M) {
  __shared__ _Float16 smem[K * 64];

  // ---- stage packed W into LDS with async copies (16B per lane per issue)
  {
    unsigned lds_base = (unsigned)(uintptr_t)(LDS3 _Float16*)smem;
#pragma unroll
    for (int c = threadIdx.x; c < K * 8; c += 128) {
      unsigned loff = lds_base + c * 16;
      unsigned long long ga = (unsigned long long)(uintptr_t)(Wpk + c * 8);
      asm volatile("global_load_async_to_lds_b128 %0, %1, off"
                   :: "v"(loff), "v"(ga) : "memory");
    }
#if defined(__has_builtin) && __has_builtin(__builtin_amdgcn_s_wait_asynccnt)
    __builtin_amdgcn_s_wait_asynccnt(0);
#else
    asm volatile("s_wait_asynccnt 0x0" ::: "memory");
#endif
    __syncthreads();
  }

  const int wave = threadIdx.x >> 5;
  const int lane = threadIdx.x & 31;
  const int tile = blockIdx.x * 4 + wave;
  if (tile * 16 >= M) return;  // wave-uniform: EXEC all-ones for WMMA

  const int kbase = (lane & 16) ? 8 : 0;
  const int mrow  = tile * 16 + (lane & 15);
  const int ncol  = lane & 15;
  const _Float16* arow = A + (size_t)mrow * K;

  v8f acc[4] = {};

  for (int k0 = 0; k0 < K; k0 += 32) {
    // A fragment: two contiguous 16B runs of pre-converted f16
    const v8h alo = *(const v8h*)(arow + k0 + kbase);
    const v8h ahi = *(const v8h*)(arow + k0 + kbase + 16);
    v16h a;
#pragma unroll
    for (int i = 0; i < 8; ++i) { a[i] = alo[i]; a[8 + i] = ahi[i]; }

    const int ks = k0 >> 5;
#pragma unroll
    for (int nt = 0; nt < 4; ++nt) {
      const v8h* bp = (const v8h*)&smem[(size_t)(((ks * 4 + nt) * 32) + lane) * 16];
      const v8h blo = bp[0], bhi = bp[1];
      v16h b;
#pragma unroll
      for (int i = 0; i < 8; ++i) { b[i] = blo[i]; b[8 + i] = bhi[i]; }
      acc[nt] = __builtin_amdgcn_wmma_f32_16x16x32_f16(
          false, a, false, b, (short)0, acc[nt], false, false);
    }
  }

  // C/D layout: VGPR r holds M = r (lanes 0-15) / 8+r (lanes 16-31)
  const int rbase = tile * 16 + ((lane & 16) ? 8 : 0);
#pragma unroll
  for (int nt = 0; nt < 4; ++nt) {
    const int n = nt * 16 + ncol;
    const float bv = bias[n];
#pragma unroll
    for (int r = 0; r < 8; ++r) {
      float val = acc[nt][r] + bv;
      if (RELU) val = fmaxf(val, 0.0f);
      C[(size_t)(rbase + r) * HID + n] = (OT)val;
    }
  }
}

// ---------------------------------------------------------------------------
// degree / normalization
// ---------------------------------------------------------------------------
__global__ void deg_init_kernel(float* __restrict__ deg, int n) {
  int i = blockIdx.x * blockDim.x + threadIdx.x;
  if (i < n) deg[i] = 1.0f;  // self-loop contribution
}
__global__ void deg_accum_kernel(const int* __restrict__ col,
                                 float* __restrict__ deg, int e) {
  int i = blockIdx.x * blockDim.x + threadIdx.x;
  if (i < e) atomic_add_f32(&deg[col[i]], 1.0f);
}
__global__ void dinv_finalize_kernel(float* __restrict__ deg, int n) {
  int i = blockIdx.x * blockDim.x + threadIdx.x;
  if (i < n) {
    float d = deg[i];
    deg[i] = (d > 0.0f) ? (1.0f / sqrtf(d)) : 0.0f;
  }
}

__global__ void zero_kernel(float* __restrict__ p, int n) {
  int i = blockIdx.x * blockDim.x + threadIdx.x;
  if (i < n) p[i] = 0.0f;
}
__global__ void relu_to_f16_kernel(const float* __restrict__ src,
                                   _Float16* __restrict__ dst, int n) {
  int i = blockIdx.x * blockDim.x + threadIdx.x;
  if (i < n) dst[i] = (_Float16)fmaxf(src[i], 0.0f);
}

// ---------------------------------------------------------------------------
// Edge attention: one thread per (edge, head); 8 channels per thread.
// kbuf/vbuf are [t][N][64] f32; scores over t<=3 history slices.
// Working set (~36MB/layer) is L2-resident on the 192MB L2.
// ---------------------------------------------------------------------------
__device__ __forceinline__ float dot4(float4 a, float4 b) {
  return a.x * b.x + a.y * b.y + a.z * b.z + a.w * b.w;
}

__global__ __launch_bounds__(256) void edge_attn_kernel(
    const int* __restrict__ row, const int* __restrict__ col,
    const float* __restrict__ dinv, const float* __restrict__ q,
    const float* __restrict__ kbuf, const float* __restrict__ vbuf,
    float* __restrict__ outacc, int E, int N, int t) {
  int gid = blockIdx.x * blockDim.x + threadIdx.x;
  int total = (E + N) * HEADS;
  if (gid >= total) return;
  const int e = gid >> 3;
  const int h = gid & 7;
  int src, dst;
  if (e < E) { src = row[e]; dst = col[e]; }
  else       { src = dst = e - E; }  // self-loop

  const float nrm = dinv[src] * dinv[dst];

  const float4* qp = (const float4*)(q + (size_t)dst * HID + h * 8);
  const float4 q0 = qp[0], q1 = qp[1];

  float scores[3];
  float smax = -3.4e38f;
  for (int tt = 0; tt < t; ++tt) {
    const float4* kp = (const float4*)(kbuf + ((size_t)tt * N + src) * HID + h * 8);
    float s = (dot4(q0, kp[0]) + dot4(q1, kp[1])) * 0.35355339059327373f;  // 1/sqrt(8)
    scores[tt] = s;
    smax = fmaxf(smax, s);
  }
  float ssum = 0.0f;
  for (int tt = 0; tt < t; ++tt) {
    scores[tt] = expf(scores[tt] - smax);
    ssum += scores[tt];
  }
  const float scale = nrm / ssum;

  float m[8] = {0, 0, 0, 0, 0, 0, 0, 0};
  for (int tt = 0; tt < t; ++tt) {
    const float w = scores[tt] * scale;
    const float4* vp = (const float4*)(vbuf + ((size_t)tt * N + src) * HID + h * 8);
    const float4 v0 = vp[0], v1 = vp[1];
    m[0] += w * v0.x; m[1] += w * v0.y; m[2] += w * v0.z; m[3] += w * v0.w;
    m[4] += w * v1.x; m[5] += w * v1.y; m[6] += w * v1.z; m[7] += w * v1.w;
  }
  float* op = outacc + (size_t)dst * HID + h * 8;
#pragma unroll
  for (int c = 0; c < 8; ++c) atomic_add_f32(op + c, m[c]);
}

// ---------------------------------------------------------------------------
// Row-wise log_softmax over 64 logits, in place.
// ---------------------------------------------------------------------------
__global__ void log_softmax_kernel(float* __restrict__ out, int M) {
  int i = blockIdx.x * blockDim.x + threadIdx.x;
  if (i >= M) return;
  float* p = out + (size_t)i * HID;
  float mx = -3.4e38f;
#pragma unroll 8
  for (int c = 0; c < HID; ++c) mx = fmaxf(mx, p[c]);
  float s = 0.0f;
#pragma unroll 8
  for (int c = 0; c < HID; ++c) s += expf(p[c] - mx);
  const float ls = logf(s) + mx;
#pragma unroll 8
  for (int c = 0; c < HID; ++c) p[c] = p[c] - ls;
}

// ---------------------------------------------------------------------------
// Orchestration
// ---------------------------------------------------------------------------
extern "C" void kernel_launch(void* const* d_in, const int* in_sizes, int n_in,
                              void* d_out, int out_size, void* d_ws, size_t ws_size,
                              hipStream_t stream) {
  const float* x      = (const float*)d_in[0];
  const int*   eidx   = (const int*)d_in[1];
  const float* lin1_w = (const float*)d_in[2];
  const float* lin1_b = (const float*)d_in[3];
  const float* wq     = (const float*)d_in[4];
  const float* bq     = (const float*)d_in[5];
  const float* wk     = (const float*)d_in[6];
  const float* bk     = (const float*)d_in[7];
  const float* wv     = (const float*)d_in[8];
  const float* bv     = (const float*)d_in[9];
  const float* lin2_w = (const float*)d_in[10];
  const float* lin2_b = (const float*)d_in[11];

  const int N = in_sizes[0] / 256;  // IN_C = 256
  const int E = in_sizes[1] / 2;    // edge_index is [2][E]
  const int* row = eidx;
  const int* col = eidx + E;

  // workspace layout (all region sizes are multiples of 16 bytes)
  char* p = (char*)d_ws;
  const size_t N64 = (size_t)N * HID;
  float*     dinv   = (float*)p;      p += (size_t)N * 4;        // N f32
  _Float16*  xh     = (_Float16*)p;   p += (size_t)N * 256 * 2;  // x in f16
  _Float16*  xall   = (_Float16*)p;   p += 4 * N64 * 2;          // 4 history slices, f16
  float*     kbuf   = (float*)p;      p += 3 * N64 * 4;
  float*     vbuf   = (float*)p;      p += 3 * N64 * 4;
  float*     qbuf   = (float*)p;      p += N64 * 4;
  float*     outacc = (float*)p;      p += N64 * 4;
  _Float16*  pkL1   = (_Float16*)p;   p += 256 * HID * 2;        // packed lin1_w
  _Float16*  pkQ    = (_Float16*)p;   p += HID * HID * 2;
  _Float16*  pkK    = (_Float16*)p;   p += HID * HID * 2;
  _Float16*  pkV    = (_Float16*)p;   p += HID * HID * 2;
  _Float16*  pkL2   = (_Float16*)p;   p += HID * HID * 2;

  const int B = 256;

  // symmetric GCN-style normalization
  deg_init_kernel<<<(N + B - 1) / B, B, 0, stream>>>(dinv, N);
  deg_accum_kernel<<<(E + B - 1) / B, B, 0, stream>>>(col, dinv, E);
  dinv_finalize_kernel<<<(N + B - 1) / B, B, 0, stream>>>(dinv, N);

  // x -> f16, pack lin1_w, then h = relu(x @ lin1_w + b) stored f16 as slice 0
  f32_to_f16_kernel<<<(N * 256 + B - 1) / B, B, 0, stream>>>(x, xh, N * 256);
  pack_w_kernel<<<(256 * HID + B - 1) / B, B, 0, stream>>>(lin1_w, pkL1, 256);
  {
    const int tiles = (N + 15) / 16;
    wmma_gemm_kernel<256, true, _Float16><<<(tiles + 3) / 4, 128, 0, stream>>>(
        xh, pkL1, lin1_b, xall, N);
  }

  for (int l = 0; l < 3; ++l) {
    const int t = l + 1;
    pack_w_kernel<<<(HID * HID + B - 1) / B, B, 0, stream>>>(
        wq + (size_t)l * HID * HID, pkQ, HID);
    pack_w_kernel<<<(HID * HID + B - 1) / B, B, 0, stream>>>(
        wk + (size_t)l * HID * HID, pkK, HID);
    pack_w_kernel<<<(HID * HID + B - 1) / B, B, 0, stream>>>(
        wv + (size_t)l * HID * HID, pkV, HID);

    // q from last slice; k,v over all t slices at once (A contiguous [t*N][64])
    {
      const int tiles = (N + 15) / 16;
      wmma_gemm_kernel<64, false, float><<<(tiles + 3) / 4, 128, 0, stream>>>(
          xall + (size_t)(t - 1) * N64, pkQ, bq + (size_t)l * HID, qbuf, N);
    }
    {
      const int Mkv   = t * N;
      const int tiles = (Mkv + 15) / 16;
      wmma_gemm_kernel<64, false, float><<<(tiles + 3) / 4, 128, 0, stream>>>(
          xall, pkK, bk + (size_t)l * HID, kbuf, Mkv);
      wmma_gemm_kernel<64, false, float><<<(tiles + 3) / 4, 128, 0, stream>>>(
          xall, pkV, bv + (size_t)l * HID, vbuf, Mkv);
    }

    zero_kernel<<<((int)N64 + B - 1) / B, B, 0, stream>>>(outacc, (int)N64);
    const int tot = (E + N) * HEADS;
    edge_attn_kernel<<<(tot + B - 1) / B, B, 0, stream>>>(
        row, col, dinv, qbuf, kbuf, vbuf, outacc, E, N, t);
    relu_to_f16_kernel<<<((int)N64 + B - 1) / B, B, 0, stream>>>(
        outacc, xall + (size_t)t * N64, (int)N64);
  }

  // logits = slice3 @ lin2_w + b ; log_softmax in place on d_out
  pack_w_kernel<<<(HID * HID + B - 1) / B, B, 0, stream>>>(lin2_w, pkL2, HID);
  {
    const int tiles = (N + 15) / 16;
    wmma_gemm_kernel<64, false, float><<<(tiles + 3) / 4, 128, 0, stream>>>(
        xall + 3 * N64, pkL2, lin2_b, (float*)d_out, N);
    log_softmax_kernel<<<(N + B - 1) / B, B, 0, stream>>>((float*)d_out, N);
  }
}
